// GCPNet_84361747628393
// MI455X (gfx1250) — matrix operand order repaired and checked by reading, not compiled, not run on hardware
//
#include <hip/hip_runtime.h>

// ---------------------------------------------------------------------------
// GCPNet on MI455X (gfx1250, wave32, WMMA + async-to-LDS).
//   EMB=64, TMAX=32, N=4096 vertex nodes, CN=512 color nodes, 32 graphs.
// GEMMs: v_wmma_f32_16x16x32_f16, double-buffered LDS staging where the A
// panel moves via global_load_async_to_lds_b128 (ASYNCcnt) and the B panel is
// written transposed into WMMA-fragment order (contiguous ds_load_b128 reads).
// Adjacency is 0/1 -> exact in f16; f32 accumulation throughout.
// LSTM recurrences: latency-bound single-workgroup scans, Whh in registers.
// ---------------------------------------------------------------------------

typedef __attribute__((ext_vector_type(16))) _Float16 v16h;
typedef __attribute__((ext_vector_type(8)))  _Float16 v8h;
typedef __attribute__((ext_vector_type(2)))  _Float16 v2h;
typedef __attribute__((ext_vector_type(8)))  float    v8f;

#define WM_RELU 1
#define WM_ACC  2

#define BM 128            // block tile rows (8 waves x 16)
#define BK 32             // K step (one WMMA k-slice)
#define SA_STRIDE 40      // halves per A row in LDS (32 data + 8 pad = 80B)
#define SB_STRIDE 24      // halves per lane in LDS B (16 data + 8 pad = 48B)

__device__ __forceinline__ unsigned lds_addr(const void* p)
{
    // Generic pointers to LDS carry the wave-relative LDS offset in addr[31:0]
    // (CDNA5 aperture rule), which is exactly what ds/async instructions take.
    return (unsigned)(uintptr_t)p;
}

// Stage one (A 128xBK, B BKx16) panel pair.
//  A: layout-preserving async copy, 2 x b128 per thread  (ASYNCcnt)
//  B: coalesced b32 read + transposing b16 LDS stores -> fragment order
__device__ __forceinline__ void stage_panels(const _Float16* __restrict__ A, int lda,
                                             const _Float16* __restrict__ B, int ldb,
                                             int m0, int n0, int k0, int t,
                                             _Float16* sA, _Float16* sB)
{
#pragma unroll
    for (int j = 0; j < 2; ++j) {
        const int c   = t * 2 + j;        // 512 x 16B chunks
        const int row = c >> 2;
        const int seg = c & 3;
        const _Float16* g = A + (size_t)(m0 + row) * lda + k0 + seg * 8;
        const unsigned  l = lds_addr(sA + row * SA_STRIDE + seg * 8);
        asm volatile("global_load_async_to_lds_b128 %0, %1, off"
                     :: "v"(l), "v"(g) : "memory");
    }
    {
        const int k = t >> 3;             // 32 rows x 8 half-pairs
        const int p = t & 7;
        const v2h w = *(const v2h*)(B + (size_t)(k0 + k) * ldb + n0 + 2 * p);
        const int L = 2 * p + ((k >> 4) << 4);   // fragment lane for col n0+2p
        const int e = k & 15;                    // fragment element (K within tile)
        sB[L * SB_STRIDE + e]       = w[0];
        sB[(L + 1) * SB_STRIDE + e] = w[1];
    }
}

// ------------------------- generic WMMA GEMM -------------------------------
// D[M,N] = act( A[M,K](f16,row-major) * B[K,N](f16,row-major)
//               + bias[N] (+ Cf if ACC) ),  M % 128 == 0, N % 16 == 0, K % 32 == 0.
// Fragment layouts (wave32); C/D per CDNA5 ISA 7.12.2, A/B K order chosen here:
//  A 16x32 f16 : lane l -> row (l&15); halves: K = 16*(l>>4)+e (contiguous, not ISA table order)
//  B 32x16 f16 : lane l -> col (l&15); halves: K = 16*(l>>4)+e (contiguous, not ISA table order)
//  C/D 16x16 f32: VGPR r -> row 8*(l>>4)+r, col (l&15)
__global__ __launch_bounds__(256) void gemm_wmma_f16(
    const _Float16* __restrict__ A, int lda,
    const _Float16* __restrict__ B, int ldb,
    float* __restrict__ Cf, _Float16* __restrict__ Ch,
    int ldc, const float* __restrict__ bias,
    int M, int N, int K, int flags)
{
    __shared__ _Float16 sA[2][BM * SA_STRIDE];
    __shared__ _Float16 sB[2][32 * SB_STRIDE];

    const int t    = threadIdx.x;
    const int lane = t & 31;
    const int wv   = t >> 5;              // wave -> m-subtile 0..7
    const int nb   = N >> 4;
    const int m0   = (blockIdx.x / nb) * BM;
    const int n0   = (blockIdx.x % nb) << 4;
    const int g    = lane >> 4;           // half-wave group
    const int lr   = lane & 15;

    v8f acc = {};
    int buf = 0;

    stage_panels(A, lda, B, ldb, m0, n0, 0, t, sA[0], sB[0]);

    for (int k0 = 0; k0 < K; k0 += BK) {
        const bool more = (k0 + BK) < K;
        if (more) {
            stage_panels(A, lda, B, ldb, m0, n0, k0 + BK, t, sA[buf ^ 1], sB[buf ^ 1]);
            asm volatile("s_wait_asynccnt 0x2" ::: "memory"); // batch k0 landed
        } else {
            asm volatile("s_wait_asynccnt 0x0" ::: "memory");
        }
        __syncthreads();                  // B ds-stores + cross-wave visibility

        const _Float16* ar = &sA[buf][(wv * 16 + lr) * SA_STRIDE + g * 16];
        const v8h a_lo = *(const v8h*)ar;
        const v8h a_hi = *(const v8h*)(ar + 8);
        const v16h a = __builtin_shufflevector(a_lo, a_hi,
                          0, 1, 2, 3, 4, 5, 6, 7, 8, 9, 10, 11, 12, 13, 14, 15);
        const _Float16* br = &sB[buf][lane * SB_STRIDE];
        const v8h b_lo = *(const v8h*)br;
        const v8h b_hi = *(const v8h*)(br + 8);
        const v16h b = __builtin_shufflevector(b_lo, b_hi,
                          0, 1, 2, 3, 4, 5, 6, 7, 8, 9, 10, 11, 12, 13, 14, 15);

        acc = __builtin_amdgcn_wmma_f32_16x16x32_f16(
                  false, a, false, b, (short)0, acc, false, false);

        __syncthreads();                  // protect buf before it is re-staged
        buf ^= 1;
    }

    const int col = n0 + lr;
    const float bv = bias ? bias[col] : 0.0f;
#pragma unroll
    for (int r = 0; r < 8; ++r) {
        const int row = m0 + wv * 16 + (g << 3) + r;
        const size_t idx = (size_t)row * ldc + col;
        float v = acc[r] + bv;
        if (flags & WM_ACC)  v += Cf[idx];
        if (flags & WM_RELU) v = fmaxf(v, 0.0f);
        if (Cf) Cf[idx] = v;
        if (Ch) Ch[idx] = (_Float16)v;
    }
}

// ------------------------- sequential LSTM scan ----------------------------
// One 256-thread workgroup. G[L,256] already holds x@Wih.T + bih + bhh.
// Thread t keeps Whh row t (64 f32) in registers; h broadcast via LDS.
// Gate order per jnp.split: [i | f | g | o].
__global__ void lstm_scan(const float* __restrict__ G,
                          const float* __restrict__ Whh,
                          float* __restrict__ st_h, float* __restrict__ st_c,
                          _Float16* __restrict__ Y, int L)
{
    __shared__ float sh_h[64];
    __shared__ float sh_g[256];
    const int t = threadIdx.x;
    float w[64];
#pragma unroll 8
    for (int j = 0; j < 64; ++j) w[j] = Whh[t * 64 + j];
    float c = 0.0f;
    if (t < 64) { sh_h[t] = st_h[t]; c = st_c[t]; }
    __syncthreads();

    for (int i = 0; i < L; ++i) {
        if (i + 8 < L) __builtin_prefetch(&G[(size_t)(i + 8) * 256 + t], 0, 1);
        float gsum = G[(size_t)i * 256 + t];
#pragma unroll 8
        for (int j = 0; j < 64; ++j) gsum += w[j] * sh_h[j];
        sh_g[t] = gsum;
        __syncthreads();
        if (t < 64) {
            const float ig = 1.0f / (1.0f + expf(-sh_g[t]));
            const float fg = 1.0f / (1.0f + expf(-sh_g[64 + t]));
            const float gg = tanhf(sh_g[128 + t]);
            const float og = 1.0f / (1.0f + expf(-sh_g[192 + t]));
            c = fg * c + ig * gg;
            const float h = og * tanhf(c);
            sh_h[t] = h;
            Y[(size_t)i * 64 + t] = (_Float16)h;
        }
        __syncthreads();
    }
    if (t < 64) { st_h[t] = sh_h[t]; st_c[t] = c; }
}

// ------------------------- final vote reduce -------------------------------
__global__ void vote_reduce(const float* __restrict__ X,
                            const float* __restrict__ W3,
                            const float* __restrict__ b3,
                            float* __restrict__ out)
{
    __shared__ float red[128];
    const int g = blockIdx.x, t = threadIdx.x;
    const float* row = X + (size_t)(g * 128 + t) * 64;
    float d = 0.0f;
#pragma unroll 8
    for (int j = 0; j < 64; ++j) d += row[j] * W3[j];
    red[t] = d;
    __syncthreads();
    for (int s = 64; s > 0; s >>= 1) {
        if (t < s) red[t] += red[t + s];
        __syncthreads();
    }
    if (t == 0) {
        const float m = red[0] * (1.0f / 128.0f) + b3[0];
        out[g] = 1.0f / (1.0f + expf(-m));
    }
}

// ------------------------- setup helpers -----------------------------------
__global__ void k_f32_to_f16(const float* __restrict__ s, _Float16* __restrict__ d, int n)
{
    for (int i = blockIdx.x * blockDim.x + threadIdx.x; i < n; i += gridDim.x * blockDim.x)
        d[i] = (_Float16)s[i];
}
// d[c*R + r] = s[r*C + c]  (W[O,I] -> W^T[I,O]; M_vc -> M_vc^T)
__global__ void k_transpose_f16(const float* __restrict__ s, _Float16* __restrict__ d,
                                int R, int C)
{
    const int n = R * C;
    for (int i = blockIdx.x * blockDim.x + threadIdx.x; i < n; i += gridDim.x * blockDim.x) {
        const int r = i / C, c = i - r * C;
        d[(size_t)c * R + r] = (_Float16)s[i];
    }
}
__global__ void k_bias_sum(const float* __restrict__ a, const float* __restrict__ b,
                           float* __restrict__ d, int n)
{
    const int i = blockIdx.x * blockDim.x + threadIdx.x;
    if (i < n) d[i] = a[i] + b[i];
}
__global__ void k_copy_f32(const float* __restrict__ s, float* __restrict__ d, int n)
{
    const int i = blockIdx.x * blockDim.x + threadIdx.x;
    if (i < n) d[i] = s[i];
}

// ------------------------- host orchestration ------------------------------
static inline void gemm(const _Float16* A, int lda, const _Float16* B, int ldb,
                        float* Cf, _Float16* Ch, int ldc, const float* bias,
                        int M, int N, int K, int flags, hipStream_t s)
{
    const int blocks = (M / BM) * (N >> 4);
    gemm_wmma_f16<<<blocks, 256, 0, s>>>(A, lda, B, ldb, Cf, Ch, ldc, bias, M, N, K, flags);
}

extern "C" void kernel_launch(void* const* d_in, const int* in_sizes, int n_in,
                              void* d_out, int out_size, void* d_ws, size_t ws_size,
                              hipStream_t stream)
{
    (void)in_sizes; (void)n_in; (void)out_size; (void)ws_size;
    constexpr int NV = 4096, NC = 512, E = 64, TMAX = 32;

    // ---- inputs (setup_inputs() dict order, tuples flattened) ----
    const float* M_vv  = (const float*)d_in[0];   // [4096,4096]
    const float* M_vc  = (const float*)d_in[1];   // [4096,512]
    const float* V0    = (const float*)d_in[2];   // [4096,64]
    const float* C0    = (const float*)d_in[3];   // [512,64]
    const float* cW[3] = {(const float*)d_in[4], (const float*)d_in[6], (const float*)d_in[8]};
    const float* cb[3] = {(const float*)d_in[5], (const float*)d_in[7], (const float*)d_in[9]};
    const float* vW[3] = {(const float*)d_in[10], (const float*)d_in[12], (const float*)d_in[14]};
    const float* vb[3] = {(const float*)d_in[11], (const float*)d_in[13], (const float*)d_in[15]};
    const float* oW[3] = {(const float*)d_in[16], (const float*)d_in[18], (const float*)d_in[20]};
    const float* ob[3] = {(const float*)d_in[17], (const float*)d_in[19], (const float*)d_in[21]};
    const float* WihV  = (const float*)d_in[22];  // [256,128]
    const float* WhhV  = (const float*)d_in[23];  // [256,64]
    const float* bihV  = (const float*)d_in[24];
    const float* bhhV  = (const float*)d_in[25];
    const float* WihC  = (const float*)d_in[26];  // [256,64]
    const float* WhhC  = (const float*)d_in[27];
    const float* bihC  = (const float*)d_in[28];
    const float* bhhC  = (const float*)d_in[29];
    const float* Vh0   = (const float*)d_in[30];
    const float* Vc0   = (const float*)d_in[31];
    const float* Ch0   = (const float*)d_in[32];
    const float* Cc0   = (const float*)d_in[33];
    float* out = (float*)d_out;                   // [32]

    // ---- workspace carve-out ----
    char*  base = (char*)d_ws;
    size_t off  = 0;
    auto take = [&](size_t bytes) -> char* {
        char* p = base + off;
        off = (off + bytes + 255) & ~(size_t)255;
        return p;
    };
    _Float16* Mvv_h = (_Float16*)take((size_t)NV * NV * 2);   // 32 MB (L2-resident)
    _Float16* Mvc_h = (_Float16*)take((size_t)NV * NC * 2);
    _Float16* Mcv_h = (_Float16*)take((size_t)NC * NV * 2);   // M_vc^T
    _Float16* Vh    = (_Float16*)take((size_t)NV * E * 2);    // current V (f16)
    _Float16* Chh   = (_Float16*)take((size_t)NC * E * 2);    // current C (f16)
    _Float16* t1c   = (_Float16*)take((size_t)NC * E * 2);
    _Float16* t2c   = (_Float16*)take((size_t)NC * E * 2);
    _Float16* CC    = (_Float16*)take((size_t)NC * E * 2);    // mlpC(C)
    _Float16* t1v   = (_Float16*)take((size_t)NV * E * 2);
    _Float16* t2v   = (_Float16*)take((size_t)NV * E * 2);
    _Float16* VVh   = (_Float16*)take((size_t)NV * E * 2);    // mlpV(V)
    _Float16* Xa    = (_Float16*)take((size_t)NV * E * 2);    // M_vv @ V
    _Float16* Xb    = (_Float16*)take((size_t)NV * E * 2);    // M_vc @ CC
    _Float16* Xc    = (_Float16*)take((size_t)NC * E * 2);    // M_vc^T @ VV
    float*    Gv    = (float*)take((size_t)NV * 256 * 4);     // vertex gate pre-act
    float*    Gc    = (float*)take((size_t)NC * 256 * 4);     // color gate pre-act
    _Float16* wc[3], *wv[3], *wo[2];
    for (int i = 0; i < 3; ++i) wc[i] = (_Float16*)take(64 * 64 * 2);
    for (int i = 0; i < 3; ++i) wv[i] = (_Float16*)take(64 * 64 * 2);
    for (int i = 0; i < 2; ++i) wo[i] = (_Float16*)take(64 * 64 * 2);
    _Float16* WihTv = (_Float16*)take(128 * 256 * 2);         // Wih_v^T [128,256]
    _Float16* WihTc = (_Float16*)take(64 * 256 * 2);          // Wih_c^T [64,256]
    float* bsv = (float*)take(256 * 4);
    float* bsc = (float*)take(256 * 4);
    float* svh = (float*)take(64 * 4);
    float* svc = (float*)take(64 * 4);
    float* sch = (float*)take(64 * 4);
    float* scc = (float*)take(64 * 4);
    float* voteF = Gv;                                        // reuse 4 MB buffer

    // ---- setup: conversions / transposes / state init (idempotent) ----
    k_f32_to_f16  <<<4096, 256, 0, stream>>>(M_vv, Mvv_h, NV * NV);
    k_f32_to_f16  <<<2048, 256, 0, stream>>>(M_vc, Mvc_h, NV * NC);
    k_transpose_f16<<<2048, 256, 0, stream>>>(M_vc, Mcv_h, NV, NC);
    k_f32_to_f16  <<<256, 256, 0, stream>>>(V0, Vh, NV * E);
    k_f32_to_f16  <<<32, 256, 0, stream>>>(C0, Chh, NC * E);
    for (int i = 0; i < 3; ++i) {
        k_transpose_f16<<<16, 256, 0, stream>>>(cW[i], wc[i], 64, 64);
        k_transpose_f16<<<16, 256, 0, stream>>>(vW[i], wv[i], 64, 64);
    }
    k_transpose_f16<<<16, 256, 0, stream>>>(oW[0], wo[0], 64, 64);
    k_transpose_f16<<<16, 256, 0, stream>>>(oW[1], wo[1], 64, 64);
    k_transpose_f16<<<128, 256, 0, stream>>>(WihV, WihTv, 256, 128);
    k_transpose_f16<<<64, 256, 0, stream>>>(WihC, WihTc, 256, 64);
    k_bias_sum<<<1, 256, 0, stream>>>(bihV, bhhV, bsv, 256);
    k_bias_sum<<<1, 256, 0, stream>>>(bihC, bhhC, bsc, 256);
    k_copy_f32<<<1, 64, 0, stream>>>(Vh0, svh, 64);
    k_copy_f32<<<1, 64, 0, stream>>>(Vc0, svc, 64);
    k_copy_f32<<<1, 64, 0, stream>>>(Ch0, sch, 64);
    k_copy_f32<<<1, 64, 0, stream>>>(Cc0, scc, 64);

    // ---- TMAX outer iterations ----
    for (int t = 0; t < TMAX; ++t) {
        // mlpC(C): 512x64, three layers (relu, relu, linear)
        gemm(Chh, E, wc[0], 64, nullptr, t1c, E, cb[0], NC, E, E, WM_RELU, stream);
        gemm(t1c, E, wc[1], 64, nullptr, t2c, E, cb[1], NC, E, E, WM_RELU, stream);
        gemm(t2c, E, wc[2], 64, nullptr, CC,  E, cb[2], NC, E, E, 0, stream);
        // mlpV(V): 4096x64
        gemm(Vh,  E, wv[0], 64, nullptr, t1v, E, vb[0], NV, E, E, WM_RELU, stream);
        gemm(t1v, E, wv[1], 64, nullptr, t2v, E, vb[1], NV, E, E, WM_RELU, stream);
        gemm(t2v, E, wv[2], 64, nullptr, VVh, E, vb[2], NV, E, E, 0, stream);
        // message GEMMs (adjacency is L2-resident in f16)
        gemm(Mvv_h, NV, Vh,  E, nullptr, Xa, E, nullptr, NV, E, NV, 0, stream);
        gemm(Mvc_h, NC, CC,  E, nullptr, Xb, E, nullptr, NV, E, NC, 0, stream);
        gemm(Mcv_h, NV, VVh, E, nullptr, Xc, E, nullptr, NC, E, NV, 0, stream);
        // gate pre-activations: G = x @ Wih^T + (bih + bhh)
        gemm(Xa, E, WihTv,            256, Gv, nullptr, 256, bsv, NV, 256, E, 0,      stream);
        gemm(Xb, E, WihTv + 64 * 256, 256, Gv, nullptr, 256, nullptr, NV, 256, E, WM_ACC, stream);
        gemm(Xc, E, WihTc,            256, Gc, nullptr, 256, bsc, NC, 256, E, 0,      stream);
        // sequential recurrences (write new V / C in f16)
        lstm_scan<<<1, 256, 0, stream>>>(Gv, WhhV, svh, svc, Vh,  NV);
        lstm_scan<<<1, 256, 0, stream>>>(Gc, WhhC, sch, scc, Chh, NC);
    }

    // ---- vote head + per-graph mean + sigmoid ----
    gemm(Vh,  E, wo[0], 64, nullptr, t1v, E, ob[0], NV, E, E, WM_RELU, stream);
    gemm(t1v, E, wo[1], 64, voteF, nullptr, E, ob[1], NV, E, E, WM_RELU, stream);
    vote_reduce<<<32, 128, 0, stream>>>(voteF, oW[2], ob[2], out);
}
